// AttentiveTransformer_29935922053780
// MI455X (gfx1250) — matrix operand order repaired and checked by reading, not compiled, not run on hardware
//
#include <hip/hip_runtime.h>

typedef __attribute__((ext_vector_type(16))) _Float16 v16h;
typedef __attribute__((ext_vector_type(8)))  float    v8f;
typedef __attribute__((ext_vector_type(4)))  float    v4f;

#define NA      128     // n_a
#define DIM     1024    // input_dim
#define VBS     128     // ghost batch size
#define BN_EPS  1e-5f
#define ZPAD    1036    // 8*1036 % 64 == 32 -> lane halves hit disjoint LDS banks

// ---------------------------------------------------------------------------
// Kernel 0: one-time W (f32 [1024][128]) -> f16 [1024][128] into workspace.
// 256 KB, L2-resident for the whole main kernel.
// ---------------------------------------------------------------------------
__global__ __launch_bounds__(256)
void convert_w_f16(const float* __restrict__ W, _Float16* __restrict__ Wh)
{
    int idx = (blockIdx.x * 256 + threadIdx.x) * 4;   // 4 elements per thread
    v4f w = *(const v4f*)(W + idx);
    _Float16 h0 = (_Float16)w.x, h1 = (_Float16)w.y,
             h2 = (_Float16)w.z, h3 = (_Float16)w.w;
    // pack as 2x u32 store via vector of 4 halves
    typedef __attribute__((ext_vector_type(4))) _Float16 v4h;
    v4h hv = { h0, h1, h2, h3 };
    *(v4h*)(Wh + idx) = hv;
}

// ---------------------------------------------------------------------------
// Kernel 1: one block per ghost-batch chunk of 128 rows. 256 threads = 8 waves.
// Fuses: GhostBatchNorm -> f16 WMMA GEMM (x @ W^T) -> *prior -> sparsemax.
// ---------------------------------------------------------------------------
__global__ __launch_bounds__(256)
void attentive_fused(const float* __restrict__ a,
                     const float* __restrict__ prior,
                     const float* __restrict__ gamma,
                     const float* __restrict__ beta,
                     const _Float16* __restrict__ Wh,
                     float* __restrict__ out)
{
    __shared__ _Float16 xh[128][136];   // normalized x tile, f16, padded rows
    __shared__ float    zs[16][ZPAD];   // z strip for one 16-row group
    __shared__ float    redS[2][128];   // partial sums
    __shared__ float    redQ[2][128];   // partial sums of squares
    __shared__ float    scaleS[128];    // rsqrt(var+eps)*gamma
    __shared__ float    shiftS[128];    // beta - mean*scale

    const int t     = threadIdx.x;
    const int chunk = blockIdx.x;
    const long rowBase = (long)chunk * VBS;

    // ---------------- Phase 1: GhostBatchNorm ----------------
    const int c    = t & 127;           // feature column
    const int half = t >> 7;            // rows [half*64, half*64+64)
    const float* ap = a + (rowBase + (long)half * 64) * NA + c;

    float s = 0.f, sq = 0.f;
    #pragma unroll 8
    for (int r = 0; r < 64; ++r) {
        float v = ap[r * NA];           // 4 waves span 128 consecutive floats
        s += v; sq += v * v;
    }
    redS[half][c] = s;
    redQ[half][c] = sq;
    __syncthreads();

    if (t < 128) {
        float sum  = redS[0][t] + redS[1][t];
        float ssq  = redQ[0][t] + redQ[1][t];
        float mean = sum * (1.f / 128.f);
        float var  = ssq * (1.f / 128.f) - mean * mean;
        float rs   = rsqrtf(var + BN_EPS);
        float g    = gamma[t];
        scaleS[t]  = rs * g;
        shiftS[t]  = beta[t] - mean * rs * g;
    }
    __syncthreads();

    {
        float sc = scaleS[c], sh = shiftS[c];
        #pragma unroll 8
        for (int r = 0; r < 64; ++r) {
            float v = ap[r * NA];       // re-read: tile hot in WGP$/L2
            xh[half * 64 + r][c] = (_Float16)(v * sc + sh);
        }
    }
    __syncthreads();

    // ---------------- Phase 2: WMMA GEMM + prior + sparsemax ----------------
    const int w    = t >> 5;            // wave 0..7
    const int lane = t & 31;
    const int lhi  = lane >> 4;         // 0/1
    const int llo  = lane & 15;

    for (int rg = 0; rg < 8; ++rg) {    // 8 groups of 16 M-rows
        // A fragments (shared 16 rows), K=0..127 in 4 chunks.
        // A k order used here (contiguous): lane<16 holds K 0..15, lane>=16 holds K 16..31 (matches B).
        v16h afr[4];
        const int m = rg * 16 + llo;
        #pragma unroll
        for (int kc = 0; kc < 4; ++kc) {
            #pragma unroll
            for (int h = 0; h < 16; ++h) {
                int k = kc * 32 + lhi * 16 + h;
                afr[kc][h] = xh[m][k];
            }
        }

        // Each wave computes 8 of the 64 N-tiles (16 cols each).
        #pragma unroll 1
        for (int i = 0; i < 8; ++i) {
            const int nt    = w + i * 8;
            const int nbase = nt * 16;
            v8f acc = {};
            #pragma unroll
            for (int kc = 0; kc < 4; ++kc) {
                // B 32x16: lane<16 -> K 0..15, lane>=16 -> K 16..31; N = lane%16.
                // 16 contiguous f16 per lane = 32B -> 2x global_load_b128, L2-hot.
                const _Float16* wp = Wh + (long)(nbase + llo) * NA + kc * 32 + lhi * 16;
                v16h bfr;
                #pragma unroll
                for (int h = 0; h < 16; ++h) bfr[h] = wp[h];
                acc = __builtin_amdgcn_wmma_f32_16x16x32_f16(
                          false, afr[kc], false, bfr, (short)0, acc, false, false);
            }
            // C/D layout: VGPR r -> (M = r + 8*lhi, N = llo). Fuse prior multiply.
            #pragma unroll
            for (int r = 0; r < 8; ++r) {
                int  mm   = r + lhi * 8;
                long grow = rowBase + rg * 16 + mm;
                float p   = prior[grow * DIM + nbase + llo];
                zs[mm][nbase + llo] = acc[r] * p;
            }
        }
        __syncthreads();

        // Sparsemax: wave w handles rows 2w, 2w+1.
        // tau solves sum(relu(z - tau)) == 1 on [max-1, max] -> bisection (exact to f32).
        #pragma unroll 1
        for (int rr = 0; rr < 2; ++rr) {
            const int row = 2 * w + rr;
            float v[32];
            float mx = -3.402823466e38f;
            #pragma unroll
            for (int j = 0; j < 32; ++j) {          // lane holds cols {lane, lane+32, ...}
                float x = zs[row][j * 32 + lane];
                v[j] = x;
                mx = fmaxf(mx, x);
            }
            #pragma unroll
            for (int off = 16; off > 0; off >>= 1)
                mx = fmaxf(mx, __shfl_xor(mx, off, 32));

            float lo = mx - 1.f, hi = mx;
            #pragma unroll 1
            for (int it = 0; it < 32; ++it) {
                float tau = 0.5f * (lo + hi);
                float sum = 0.f;
                #pragma unroll
                for (int j = 0; j < 32; ++j) sum += fmaxf(v[j] - tau, 0.f);
                #pragma unroll
                for (int off = 16; off > 0; off >>= 1)
                    sum += __shfl_xor(sum, off, 32);
                if (sum > 1.f) lo = tau; else hi = tau;
            }
            const float tau = 0.5f * (lo + hi);

            const long grow = rowBase + rg * 16 + row;
            float* op = out + grow * DIM;
            #pragma unroll
            for (int j = 0; j < 32; ++j)            // per j: 32 lanes write 128B contiguous
                op[j * 32 + lane] = fmaxf(v[j] - tau, 0.f);
        }
        __syncthreads();
    }
}

extern "C" void kernel_launch(void* const* d_in, const int* in_sizes, int n_in,
                              void* d_out, int out_size, void* d_ws, size_t ws_size,
                              hipStream_t stream) {
    const float* a     = (const float*)d_in[0];
    const float* prior = (const float*)d_in[1];
    const float* gamma = (const float*)d_in[2];
    const float* beta  = (const float*)d_in[3];
    const float* W     = (const float*)d_in[4];
    float* out = (float*)d_out;
    _Float16* Wh = (_Float16*)d_ws;      // 1024*128 f16 = 256 KB

    // W -> f16 (DIM*NA = 131072 elems, 4 per thread, 256 threads/block -> 128 blocks)
    convert_w_f16<<<dim3((DIM * NA) / (256 * 4)), 256, 0, stream>>>(W, Wh);

    const int B = in_sizes[0] / NA;      // 65536
    attentive_fused<<<dim3(B / VBS), 256, 0, stream>>>(a, prior, gamma, beta, Wh, out);
}
